// FreeDAMasker_66992899883436
// MI455X (gfx1250) — matrix operand order repaired and checked by reading, not compiled, not run on hardware
//
#include <hip/hip_runtime.h>
#include <hip/hip_bf16.h>

typedef __attribute__((ext_vector_type(16))) _Float16 v16h;
typedef __attribute__((ext_vector_type(8)))  float    v8f;
typedef int v4i __attribute__((vector_size(16)));      // matches builtin param type

// Problem geometry (fixed by the reference):
//   image_feat [B=8, C=512, H=32, W=32]  -> M = 8192 pixels
//   proto_emb  [N=150, K=32, C=512]      -> 4800 prototype rows
//   out: soft_mask [8,150,32,32] then sim [8,150,32,32] (concatenated)
#define MPIX      8192
#define CDIM      512
#define NPROTO    150
#define KPROTO    32
#define OUT_ONE   1228800          // 8*150*32*32
#define LDS_PITCH 40               // 32 halves + 8 pad -> 20-dword row stride (bank-conflict-free b128)

// ---- CDNA5 async global->LDS path (probe-guarded; falls back to plain loads) ----
#define AS1 __attribute__((address_space(1)))
#define AS3 __attribute__((address_space(3)))
#if defined(__AMDGCN__) && __has_builtin(__builtin_amdgcn_global_load_async_to_lds_b128)
#define HAVE_ASYNC_LDS 1
#else
#define HAVE_ASYNC_LDS 0
#endif
#if defined(__AMDGCN__) && __has_builtin(__builtin_amdgcn_s_wait_asynccnt)
#define WAIT_ASYNC() __builtin_amdgcn_s_wait_asynccnt(0)
#else
#define WAIT_ASYNC() asm volatile("s_wait_asynccnt 0x0" ::: "memory")
#endif

// ---------------- Kernel 1: L2-normalize image along C, emit f16 X[M][C] ----------------
__global__ void norm_img_kernel(const float* __restrict__ img, _Float16* __restrict__ Xn) {
    int mg = blockIdx.x * blockDim.x + threadIdx.x;      // pixel id 0..8191
    int b = mg >> 10;                                    // B index
    int p = mg & 1023;                                   // h*32+w
    const float* base = img + ((size_t)b << 19) + p;     // b*512*1024 + p
    float s = 0.0f;
    #pragma unroll 8
    for (int c = 0; c < CDIM; ++c) {
        float v = base[c << 10];                         // stride H*W = 1024 (coalesced across threads)
        s += v * v;
    }
    float inv = rsqrtf(s + 1e-12f);
    _Float16* dst = Xn + (size_t)mg * CDIM;
    #pragma unroll 8
    for (int c = 0; c < CDIM; ++c)
        dst[c] = (_Float16)(base[c << 10] * inv);
}

// ---------------- Kernel 2: L2-normalize proto rows, emit f16 P[4800][512] ----------------
__global__ void norm_proto_kernel(const float* __restrict__ pro, _Float16* __restrict__ Pn) {
    int wid  = (blockIdx.x * blockDim.x + threadIdx.x) >> 5;  // one wave per row
    int lane = threadIdx.x & 31;
    if (wid >= NPROTO * KPROTO) return;
    const float* src = pro + (size_t)wid * CDIM;
    float s = 0.0f;
    #pragma unroll
    for (int i = 0; i < 16; ++i) {
        float v = src[i * 32 + lane];
        s += v * v;
    }
    #pragma unroll
    for (int d = 1; d <= 16; d <<= 1) s += __shfl_xor(s, d, 32);
    float inv = rsqrtf(s + 1e-12f);
    _Float16* dst = Pn + (size_t)wid * CDIM;
    #pragma unroll
    for (int i = 0; i < 16; ++i)
        dst[i * 32 + lane] = (_Float16)(src[i * 32 + lane] * inv);
}

// ---------------- Kernel 3: fused WMMA GEMM + K-group max/mean + sigmoid ----------------
// Grid: x = M/256 (=32), y = 75 (2 prototypes per WG). Block = 256 threads = 8 waves.
// Each wave: 32 rows x 64 cols (2 prototypes' K-columns) = 8 WMMA tiles, f32 accum,
// then shfl_xor reduce per prototype over 32 columns -> sim, sigmoid -> mask.
__global__ __launch_bounds__(256)
void simmap_wmma_kernel(const _Float16* __restrict__ Xn,
                        const _Float16* __restrict__ Pn,
                        float* __restrict__ out_mask,
                        float* __restrict__ out_sim) {
    __shared__ __align__(16) _Float16 ldsA[256 * LDS_PITCH];   // 20.0 KB
    __shared__ __align__(16) _Float16 ldsB[64  * LDS_PITCH];   //  5.0 KB

    const int t        = threadIdx.x;
    const int wave     = t >> 5;
    const int lane     = t & 31;
    const int m        = lane & 15;      // row within A tile / col within B,D tile
    const int h        = lane >> 4;      // lane half
    const int protoPair = blockIdx.y;    // 0..74, two prototypes each
    const int rowBase   = blockIdx.x * 256;
    const int protoRow  = protoPair * 64;          // row into Pn

    union Frag { v16h v; uint4 q[2]; };

    v8f acc[2][4];                        // [row-half 16][column tile of 16]
    #pragma unroll
    for (int r = 0; r < 2; ++r)
        #pragma unroll
        for (int ct = 0; ct < 4; ++ct) acc[r][ct] = (v8f){};

    for (int k0 = 0; k0 < CDIM; k0 += 32) {
        // -------- cooperative LDS fill: thread t owns one 64B row-segment --------
        {
            const _Float16* srcA = Xn + (size_t)(rowBase + t) * CDIM + k0;
            _Float16*       dstA = ldsA + t * LDS_PITCH;
            const _Float16* srcB = Pn + (size_t)(protoRow + t) * CDIM + k0;  // valid only t<64
            _Float16*       dstB = ldsB + t * LDS_PITCH;
#if HAVE_ASYNC_LDS
            // GLOBAL_LOAD_ASYNC_TO_LDS_B128: INST_OFFSET is added to both the
            // global and LDS addresses, so one base pair + imm offsets covers 64B.
            __builtin_amdgcn_global_load_async_to_lds_b128((AS1 v4i*)srcA, (AS3 v4i*)dstA,  0, 0);
            __builtin_amdgcn_global_load_async_to_lds_b128((AS1 v4i*)srcA, (AS3 v4i*)dstA, 16, 0);
            __builtin_amdgcn_global_load_async_to_lds_b128((AS1 v4i*)srcA, (AS3 v4i*)dstA, 32, 0);
            __builtin_amdgcn_global_load_async_to_lds_b128((AS1 v4i*)srcA, (AS3 v4i*)dstA, 48, 0);
            if (t < 64) {
                __builtin_amdgcn_global_load_async_to_lds_b128((AS1 v4i*)srcB, (AS3 v4i*)dstB,  0, 0);
                __builtin_amdgcn_global_load_async_to_lds_b128((AS1 v4i*)srcB, (AS3 v4i*)dstB, 16, 0);
                __builtin_amdgcn_global_load_async_to_lds_b128((AS1 v4i*)srcB, (AS3 v4i*)dstB, 32, 0);
                __builtin_amdgcn_global_load_async_to_lds_b128((AS1 v4i*)srcB, (AS3 v4i*)dstB, 48, 0);
            }
            __builtin_prefetch((const char*)srcA + 64, 0, 3);   // next K-step into near cache
            WAIT_ASYNC();
#else
            {
                const uint4* s4 = (const uint4*)srcA;
                uint4* d4 = (uint4*)dstA;
                d4[0] = s4[0]; d4[1] = s4[1]; d4[2] = s4[2]; d4[3] = s4[3];
                __builtin_prefetch((const char*)srcA + 64, 0, 3);
                if (t < 64) {
                    const uint4* s4b = (const uint4*)srcB;
                    uint4* d4b = (uint4*)dstB;
                    d4b[0] = s4b[0]; d4b[1] = s4b[1]; d4b[2] = s4b[2]; d4b[3] = s4b[3];
                    __builtin_prefetch((const char*)srcB + 64, 0, 3);
                }
            }
#endif
        }
        __syncthreads();

        // -------- fragments (ISA 16-bit A 16x32 / dense B 32x16 layouts) --------
        Frag a[2];
        #pragma unroll
        for (int r = 0; r < 2; ++r) {
            const _Float16* arow = ldsA + (wave * 32 + r * 16 + m) * LDS_PITCH;
            a[r].q[0] = *(const uint4*)(arow + h * 8);        // K = h*8   .. h*8+7
            a[r].q[1] = *(const uint4*)(arow + 16 + h * 8);   // K = 16+h*8 ..
        }
        #pragma unroll
        for (int ct = 0; ct < 4; ++ct) {
            Frag b;
            const _Float16* brow = ldsB + (ct * 16 + m) * LDS_PITCH;
            b.q[0] = *(const uint4*)(brow + h * 8);           // K = h*8 .. h*8+7
            b.q[1] = *(const uint4*)(brow + 16 + h * 8);      // K = 16+h*8 ..
            #pragma unroll
            for (int r = 0; r < 2; ++r)
                acc[r][ct] = __builtin_amdgcn_wmma_f32_16x16x32_f16(
                    false, a[r].v, false, b.v, (short)0, acc[r][ct], false, false);
        }
        __syncthreads();
    }

    // -------- fused reduction: per prototype (pairs of column tiles) --------
    // C/D layout: lanes 0-15 -> rows v (N = lane), lanes 16-31 -> rows v+8 (N = lane-16).
    // xor masks 1,2,4,8 reduce across the 16 columns held by each lane-half.
    #pragma unroll
    for (int pp = 0; pp < 2; ++pp) {
        #pragma unroll
        for (int r = 0; r < 2; ++r) {
            #pragma unroll
            for (int v = 0; v < 8; ++v) {
                float x0 = acc[r][2 * pp][v];
                float x1 = acc[r][2 * pp + 1][v];
                float mx = fmaxf(x0, x1);
                float sm = x0 + x1;
                #pragma unroll
                for (int d = 1; d <= 8; d <<= 1) {
                    mx = fmaxf(mx, __shfl_xor(mx, d, 32));
                    sm += __shfl_xor(sm, d, 32);
                }
                if ((lane & 15) == 0) {                 // lanes 0 and 16 write
                    int mg  = rowBase + wave * 32 + r * 16 + v + h * 8;
                    int bb  = mg >> 10;
                    int p   = mg & 1023;
                    int nP  = protoPair * 2 + pp;
                    int idx = bb * (NPROTO * 1024) + nP * 1024 + p;   // [B,N,H,W]
                    float sim = 0.5f * mx + sm * (0.5f / 32.0f);      // 0.5*max + 0.5*mean
                    out_mask[idx] = 1.0f / (1.0f + __expf(-sim));
                    out_sim[idx]  = sim;
                }
            }
        }
    }
}

extern "C" void kernel_launch(void* const* d_in, const int* in_sizes, int n_in,
                              void* d_out, int out_size, void* d_ws, size_t ws_size,
                              hipStream_t stream) {
    const float* image_feat = (const float*)d_in[0];   // [8,512,32,32] f32
    const float* proto_emb  = (const float*)d_in[1];   // [150,32,512]  f32
    float* out = (float*)d_out;                        // mask | sim, each 1228800 f32

    // Workspace: f16 normalized operands
    _Float16* Xn = (_Float16*)d_ws;                                    // 8192*512*2  = 8 MB
    _Float16* Pn = (_Float16*)((char*)d_ws + (size_t)MPIX * CDIM * 2); // 4800*512*2 = 4.8 MB

    norm_img_kernel<<<MPIX / 256, 256, 0, stream>>>(image_feat, Xn);
    norm_proto_kernel<<<(NPROTO * KPROTO * 32) / 256, 256, 0, stream>>>(proto_emb, Pn);

    dim3 grid(MPIX / 256, NPROTO / 2);
    simmap_wmma_kernel<<<grid, 256, 0, stream>>>(Xn, Pn, out, out + OUT_ONE);
}
